// NMFD_GNN_52639119180193
// MI455X (gfx1250) — compile-verified, matrix operand order checked by reading, not run on hardware
//
#include <hip/hip_runtime.h>
#include <hip/hip_bf16.h>

typedef __attribute__((ext_vector_type(16))) __bf16 v16bf;
typedef __attribute__((ext_vector_type(8)))  float  v8f;

#define NB 32
#define NSN 256
#define MT 48
#define G2D 64
#define HIDD 64
#define PDIM 10
#define BNM (NB*MT)           // 1536
#define ROWS (BNM*NSN)        // 393216
#define SEQ (NSN*NB)          // 8192

static __device__ __forceinline__ unsigned short f2bf(float x) {
  unsigned u = __float_as_uint(x);
  unsigned r = u + 0x7FFFu + ((u >> 16) & 1u);
  return (unsigned short)(r >> 16);
}
static __device__ __forceinline__ float bf2f(unsigned short h) {
  return __uint_as_float(((unsigned)h) << 16);
}
static __device__ __forceinline__ float sigm(float x) { return 1.0f / (1.0f + expf(-x)); }

// WMMA 16-bit operand tile with CONTIGUOUS K (stride_k == 1): per lane the 16
// elements are two contiguous 8-element K-runs -> two b128 loads.
// lane<16: K kh..kh+7 & kh+16..kh+23 with kh=0; lane>=16: kh=8.
static __device__ __forceinline__ v16bf load_tile_k1(const __bf16* p, int stride_m) {
  int lane = threadIdx.x & 31;
  const __bf16* q = p + (size_t)(lane & 15) * stride_m + ((lane >> 4) << 3);
  union { uint4 u[2]; v16bf v; } u;
  u.u[0] = *(const uint4*)q;
  u.u[1] = *(const uint4*)(q + 16);
  return u.v;
}

static __device__ __forceinline__ v8f wmma_bf16(v16bf a, v16bf b, v8f c) {
  return __builtin_amdgcn_wmma_f32_16x16x32_bf16(false, a, false, b, (short)0, c, false, false);
}

static __device__ __forceinline__ uint4 pack8bf(v8f a) {
  union { unsigned short s[8]; uint4 u; } p;
#pragma unroll
  for (int r = 0; r < 8; ++r) p.s[r] = f2bf(a[r]);
  return p.u;
}

// ---------- adjacency prep ----------
__global__ void k_deg(const float* __restrict__ adj, float* __restrict__ dinv) {
  int s = threadIdx.x;
  float sum = 1.0f;                       // + identity diag
  for (int t = 0; t < NSN; ++t) sum += adj[s * NSN + t];
  dinv[s] = rsqrtf(sum);
}

__global__ void k_dad(const float* __restrict__ adj, const float* __restrict__ dinv,
                      unsigned short* __restrict__ dad) {
  int s = blockIdx.x, t = threadIdx.x;
  float v = adj[s * NSN + t] + (s == t ? 1.0f : 0.0f);
  dad[s * NSN + t] = f2bf(dinv[s] * v * dinv[t]);
}

__global__ void k_adp(const float* __restrict__ L1, const float* __restrict__ L2,
                      unsigned short* __restrict__ aadp) {
  __shared__ float red[NSN];
  int t = blockIdx.x, s = threadIdx.x;
  float r = 0.0f;
  for (int p = 0; p < PDIM; ++p) r += L1[s * PDIM + p] * L2[p * NSN + t];
  r = fmaxf(r, 0.0f);
  red[s] = r; __syncthreads();
  for (int o = NSN / 2; o > 0; o >>= 1) { if (s < o) red[s] = fmaxf(red[s], red[s + o]); __syncthreads(); }
  float mx = red[0]; __syncthreads();
  float e = expf(r - mx);
  red[s] = e; __syncthreads();
  for (int o = NSN / 2; o > 0; o >>= 1) { if (s < o) red[s] += red[s + o]; __syncthreads(); }
  aadp[s * NSN + t] = f2bf(e / red[0]);   // softmax over rows (axis=0) per column t
}

// ---------- weight f32 -> bf16 ----------
__global__ void k_cvt(const float* __restrict__ src, unsigned short* __restrict__ dst, int n) {
  int i = blockIdx.x * blockDim.x + threadIdx.x;
  if (i < n) dst[i] = f2bf(src[i]);
}

// 64x64 transpose-convert: dst[g][k] = src[k][g]  (for W2/W2_adp -> contiguous-K B operand)
__global__ void k_cvt_t64(const float* __restrict__ src, unsigned short* __restrict__ dst) {
  int idx = blockIdx.x * 256 + threadIdx.x;   // 4096 total
  int g = idx >> 6, k = idx & 63;
  dst[idx] = f2bf(src[k * 64 + g]);
}

// ---------- GCN layer-1 feature matmul (K=2, pure VALU) ----------
// writes TRANSPOSED panels: buf[b][g(64)][s(256)]  (B operand with contiguous K=s)
__global__ void k_xw1(const float* __restrict__ xf, const float* __restrict__ xo,
                      const float* __restrict__ W1, const float* __restrict__ W1a,
                      unsigned short* __restrict__ bfix, unsigned short* __restrict__ badp) {
  size_t idx = (size_t)blockIdx.x * blockDim.x + threadIdx.x;
  if (idx >= (size_t)ROWS * G2D) return;
  int s  = (int)(idx & 255);
  int g  = (int)((idx >> 8) & 63);
  int bm = (int)(idx >> 14);
  int m = bm % MT, n = bm / MT;
  float x0 = xf[((size_t)n * NSN + s) * MT + m];
  float x1 = xo[((size_t)n * NSN + s) * MT + m];
  size_t o = (size_t)bm * (G2D * NSN) + (size_t)g * NSN + s;   // coalesced over s
  bfix[o] = f2bf(x0 * W1[g]  + x1 * W1[G2D + g]);
  badp[o] = f2bf(x0 * W1a[g] + x1 * W1a[G2D + g]);
}

// ---------- GCN aggregation: out = tanh(DAD@B) + tanh(Aadp@Badp), 16x64 tile/wave ----------
// B panels in [b][g][s] layout; output X in [b][s][g] (A operand for next stage).
__global__ void k_gcn_spmm(const unsigned short* __restrict__ dad, const unsigned short* __restrict__ aadp,
                           const unsigned short* __restrict__ bfix, const unsigned short* __restrict__ badp,
                           unsigned short* __restrict__ out) {
  __shared__ __align__(16) unsigned short otile[16 * G2D];
  int s0 = blockIdx.x * 16;
  int b  = blockIdx.y;
  const __bf16* Af = (const __bf16*)dad  + (size_t)s0 * NSN;
  const __bf16* Aa = (const __bf16*)aadp + (size_t)s0 * NSN;
  const __bf16* Bf = (const __bf16*)bfix + (size_t)b * NSN * G2D;
  const __bf16* Ba = (const __bf16*)badp + (size_t)b * NSN * G2D;
  v8f accF[4] = {}; v8f accA[4] = {};
  for (int k0 = 0; k0 < NSN; k0 += 32) {
    v16bf af = load_tile_k1(Af + k0, NSN);
    v16bf aa = load_tile_k1(Aa + k0, NSN);
#pragma unroll
    for (int nt = 0; nt < 4; ++nt) {
      v16bf bf = load_tile_k1(Bf + (size_t)(nt * 16) * NSN + k0, NSN);
      accF[nt] = wmma_bf16(af, bf, accF[nt]);
      v16bf ba = load_tile_k1(Ba + (size_t)(nt * 16) * NSN + k0, NSN);
      accA[nt] = wmma_bf16(aa, ba, accA[nt]);
    }
  }
  int lane = threadIdx.x & 31, nc = lane & 15, mb = (lane >> 4) * 8;
#pragma unroll
  for (int nt = 0; nt < 4; ++nt)
#pragma unroll
    for (int r = 0; r < 8; ++r)
      otile[(mb + r) * G2D + nt * 16 + nc] = f2bf(tanhf(accF[nt][r]) + tanhf(accA[nt][r]));
  __syncthreads();
  for (int c = lane; c < 128; c += 32) {          // 16 rows x 8 chunks of 8 bf16
    int row = c >> 3, part = c & 7;
    *(uint4*)&out[((size_t)b * NSN + s0 + row) * G2D + part * 8] =
        *(const uint4*)&otile[row * G2D + part * 8];
  }
}

// ---------- GCN layer-2 feature GEMM: X1 @ W2 / X1 @ W2_adp (K=N=64) ----------
// X in [b][s][g]; W2t transposed [g_out][k]; outputs TRANSPOSED panels [b][g][s].
__global__ void k_xw2(const unsigned short* __restrict__ X, const unsigned short* __restrict__ W2t,
                      const unsigned short* __restrict__ W2at,
                      unsigned short* __restrict__ bfix, unsigned short* __restrict__ badp) {
  size_t r0 = (size_t)blockIdx.x * 16;
  int b  = (int)(r0 >> 8);
  int s0 = (int)(r0 & 255);
  const __bf16* A = (const __bf16*)X + r0 * G2D;
  v8f accF[4] = {}, accA[4] = {};
#pragma unroll
  for (int k0 = 0; k0 < G2D; k0 += 32) {
    v16bf a = load_tile_k1(A + k0, G2D);
#pragma unroll
    for (int nt = 0; nt < 4; ++nt) {
      v16bf bf = load_tile_k1((const __bf16*)W2t  + (size_t)(nt * 16) * G2D + k0, G2D);
      accF[nt] = wmma_bf16(a, bf, accF[nt]);
      v16bf ba = load_tile_k1((const __bf16*)W2at + (size_t)(nt * 16) * G2D + k0, G2D);
      accA[nt] = wmma_bf16(a, ba, accA[nt]);
    }
  }
  int lane = threadIdx.x & 31, nc = lane & 15, mb = (lane >> 4) * 8;
#pragma unroll
  for (int nt = 0; nt < 4; ++nt) {
    size_t o = (size_t)b * (G2D * NSN) + (size_t)(nt * 16 + nc) * NSN + s0 + mb;
    *(uint4*)&bfix[o] = pack8bf(accF[nt]);      // 8 consecutive s per lane -> b128
    *(uint4*)&badp[o] = pack8bf(accA[nt]);
  }
}

// ---------- persistent BiLSTM direction kernel: 16 sequences per wave ----------
__global__ void k_lstm(const unsigned short* __restrict__ x, const unsigned short* __restrict__ Wih,
                       const unsigned short* __restrict__ Whh,
                       const float* __restrict__ bih, const float* __restrict__ bhh,
                       unsigned short* __restrict__ Hout, int Kin, int layer) {
  __shared__ __align__(16) unsigned short h_lds[16 * HIDD];  // h state bf16 [m][64]
  __shared__ __align__(16) float gates_t[256 * 16];          // [col][m] (b128-friendly)
  int dir = blockIdx.y;
  int seq0 = blockIdx.x * 16;
  int lane = threadIdx.x & 31, nc = lane & 15, mb = (lane >> 4) * 8;

  const __bf16* Wihd = (const __bf16*)Wih + (size_t)dir * 256 * Kin;
  const __bf16* Whhd = (const __bf16*)Whh + (size_t)dir * 256 * HIDD;
  const float* bihd = bih + dir * 256;
  const float* bhhd = bhh + dir * 256;

  float bI[4], bF[4], bG[4], bO[4];
#pragma unroll
  for (int nt = 0; nt < 4; ++nt) {
    int hx = nt * 16 + nc;
    bI[nt] = bihd[hx]       + bhhd[hx];
    bF[nt] = bihd[64 + hx]  + bhhd[64 + hx];
    bG[nt] = bihd[128 + hx] + bhhd[128 + hx];
    bO[nt] = bihd[192 + hx] + bhhd[192 + hx];
  }

  size_t xbase, rowStride, tStride;
  if (layer == 0) {           // x = GCN out [n][t][s][64]; seq = s*NB + n
    rowStride = (size_t)MT * NSN * G2D;
    tStride   = (size_t)NSN * G2D;
    xbase = (size_t)(seq0 & 31) * rowStride + (size_t)(seq0 >> 5) * G2D;
  } else {                    // x = H1 [seq][t][128]
    rowStride = (size_t)MT * 128;
    tStride   = 128;
    xbase = (size_t)seq0 * rowStride;
  }

  for (int i = threadIdx.x; i < 16 * HIDD; i += 32) h_lds[i] = 0;
  __syncthreads();

  v8f creg[4] = {};           // cell state in C-matrix register layout

#pragma unroll 1
  for (int step = 0; step < MT; ++step) {
    int t = dir ? (MT - 1 - step) : step;
    const __bf16* xrow = (const __bf16*)x + xbase + (size_t)t * tStride;
#pragma unroll 1
    for (int chunk = 0; chunk < 4; ++chunk) {   // chunk == gate (i,f,g,o), 64 cols each
      v8f acc[4] = {};
      for (int k0 = 0; k0 < Kin; k0 += 32) {    // x_t @ Wih^T
        v16bf a = load_tile_k1(xrow + k0, (int)rowStride);
#pragma unroll
        for (int nt = 0; nt < 4; ++nt) {
          int nb = chunk * 64 + nt * 16;
          v16bf b = load_tile_k1(Wihd + (size_t)nb * Kin + k0, Kin);
          acc[nt] = wmma_bf16(a, b, acc[nt]);
        }
      }
#pragma unroll
      for (int k0 = 0; k0 < HIDD; k0 += 32) {   // h @ Whh^T
        v16bf a = load_tile_k1((const __bf16*)h_lds + k0, HIDD);
#pragma unroll
        for (int nt = 0; nt < 4; ++nt) {
          int nb = chunk * 64 + nt * 16;
          v16bf b = load_tile_k1(Whhd + (size_t)nb * HIDD + k0, HIDD);
          acc[nt] = wmma_bf16(a, b, acc[nt]);
        }
      }
#pragma unroll
      for (int nt = 0; nt < 4; ++nt) {          // spill to LDS as two b128 per tile
        int col = chunk * 64 + nt * 16 + nc;
        float* g0 = &gates_t[col * 16 + mb];
        *(float4*)g0       = make_float4(acc[nt][0], acc[nt][1], acc[nt][2], acc[nt][3]);
        *(float4*)(g0 + 4) = make_float4(acc[nt][4], acc[nt][5], acc[nt][6], acc[nt][7]);
      }
    }
    __syncthreads();
#pragma unroll 1
    for (int nt = 0; nt < 4; ++nt) {
      int hx = nt * 16 + nc;
      __align__(16) float gi8[8], gf8[8], gg8[8], go8[8];
      *(float4*)&gi8[0] = *(const float4*)&gates_t[(hx)*16 + mb];
      *(float4*)&gi8[4] = *(const float4*)&gates_t[(hx)*16 + mb + 4];
      *(float4*)&gf8[0] = *(const float4*)&gates_t[(64 + hx)*16 + mb];
      *(float4*)&gf8[4] = *(const float4*)&gates_t[(64 + hx)*16 + mb + 4];
      *(float4*)&gg8[0] = *(const float4*)&gates_t[(128 + hx)*16 + mb];
      *(float4*)&gg8[4] = *(const float4*)&gates_t[(128 + hx)*16 + mb + 4];
      *(float4*)&go8[0] = *(const float4*)&gates_t[(192 + hx)*16 + mb];
      *(float4*)&go8[4] = *(const float4*)&gates_t[(192 + hx)*16 + mb + 4];
#pragma unroll
      for (int r = 0; r < 8; ++r) {
        float cn = sigm(gf8[r] + bF[nt]) * creg[nt][r] + sigm(gi8[r] + bI[nt]) * tanhf(gg8[r] + bG[nt]);
        creg[nt][r] = cn;
        float h = sigm(go8[r] + bO[nt]) * tanhf(cn);
        h_lds[(mb + r) * HIDD + hx] = f2bf(h);
      }
    }
    __syncthreads();
    for (int c = lane; c < 128; c += 32) {      // cooperative b128 store of h tile
      int row = c >> 3, part = c & 7;
      *(uint4*)&Hout[((size_t)(seq0 + row) * MT + t) * 128 + dir * 64 + part * 8] =
          *(const uint4*)&h_lds[row * HIDD + part * 8];
    }
    __syncthreads();
  }
}

// ---------- output MLP (128 -> 2) ----------
__global__ void k_mlp(const unsigned short* __restrict__ H2, const float* __restrict__ Wout,
                      float* __restrict__ out) {
  size_t idx = (size_t)blockIdx.x * blockDim.x + threadIdx.x;
  if (idx >= (size_t)NB * NSN * MT) return;
  int t = (int)(idx % MT);
  size_t ns_ = idx / MT;
  int s = (int)(ns_ % NSN);
  int n = (int)(ns_ / NSN);
  const unsigned short* h = H2 + (((size_t)s * NB + n) * MT + t) * 128;
  float f = 0.0f, o = 0.0f;
  for (int k8 = 0; k8 < 128; k8 += 8) {
    union { uint4 u; unsigned short s[8]; } w;
    w.u = *(const uint4*)&h[k8];
#pragma unroll
    for (int j = 0; j < 8; ++j) {
      float v = bf2f(w.s[j]);
      f += v * Wout[(k8 + j) * 2 + 0];
      o += v * Wout[(k8 + j) * 2 + 1];
    }
  }
  out[idx] = f;                           // flow_y (n,ns,m)
  out[(size_t)NB * NSN * MT + idx] = o;   // occ_y
}

// ---------- q_hat / q_hat_theta reductions ----------
__global__ void k_qhat(const float* __restrict__ io, const float* __restrict__ slen,
                       const float* __restrict__ theta, float* __restrict__ out) {
  __shared__ float rq[NSN], rk[NSN], rl[NSN];
  int b = blockIdx.x;
  int n = b / MT, t = b % MT;
  int s = threadIdx.x;
  float len = slen[s];
  float fl = io[((size_t)n * NSN + s) * MT + t];
  float oc = io[(size_t)NB * NSN * MT + ((size_t)n * NSN + s) * MT + t];
  float forig = fl * 100.0f + 300.0f;
  float kk = (oc * 0.05f + 0.1f) / 0.15f;
  rq[s] = forig * len; rk[s] = kk * len; rl[s] = len;
  __syncthreads();
  for (int o = NSN / 2; o > 0; o >>= 1) {
    if (s < o) { rq[s] += rq[s + o]; rk[s] += rk[s + o]; rl[s] += rl[s + o]; }
    __syncthreads();
  }
  if (s == 0) {
    float qh = rq[0] / rl[0];
    float kh = rk[0] / rl[0];
    float e0 = expf(theta[0]), e1 = expf(theta[1]), e2 = expf(theta[2]),
          e3 = expf(theta[3]), e4 = expf(theta[4]);
    float qt = -e0 * logf(expf(-e1 * kh) + e2 + expf(-e3 + e4 * kh));
    size_t base = (size_t)2 * NB * NSN * MT;
    out[base + (size_t)n * MT + t] = qh;
    out[base + (size_t)NB * MT + (size_t)n * MT + t] = qt;
  }
}

extern "C" void kernel_launch(void* const* d_in, const int* in_sizes, int n_in,
                              void* d_out, int out_size, void* d_ws, size_t ws_size,
                              hipStream_t stream) {
  (void)in_sizes; (void)n_in; (void)out_size; (void)ws_size;
  const float* x_f  = (const float*)d_in[0];
  const float* x_o  = (const float*)d_in[1];
  const float* adj  = (const float*)d_in[2];
  const float* slen = (const float*)d_in[3];
  const float* W1   = (const float*)d_in[4];
  const float* W1a  = (const float*)d_in[5];
  const float* W2   = (const float*)d_in[6];
  const float* W2a  = (const float*)d_in[7];
  const float* L1   = (const float*)d_in[8];
  const float* L2   = (const float*)d_in[9];
  const float* Wih0 = (const float*)d_in[10];
  const float* Whh0 = (const float*)d_in[11];
  const float* bih0 = (const float*)d_in[12];
  const float* bhh0 = (const float*)d_in[13];
  const float* Wih1 = (const float*)d_in[14];
  const float* Whh1 = (const float*)d_in[15];
  const float* bih1 = (const float*)d_in[16];
  const float* bhh1 = (const float*)d_in[17];
  const float* Wout = (const float*)d_in[18];
  const float* theta= (const float*)d_in[19];
  float* out = (float*)d_out;

  char* ws = (char*)d_ws;
  float* dinv          = (float*)ws;
  unsigned short* dad  = (unsigned short*)(ws + 1024);
  unsigned short* aadp = (unsigned short*)(ws + 1024 + 131072);
  size_t woff = 1024 + 2 * 131072;
  unsigned short* W2t   = (unsigned short*)(ws + woff); woff += 4096 * 2;
  unsigned short* W2at  = (unsigned short*)(ws + woff); woff += 4096 * 2;
  unsigned short* Wih0b = (unsigned short*)(ws + woff); woff += 32768 * 2;
  unsigned short* Whh0b = (unsigned short*)(ws + woff); woff += 32768 * 2;
  unsigned short* Wih1b = (unsigned short*)(ws + woff); woff += 65536 * 2;
  unsigned short* Whh1b = (unsigned short*)(ws + woff); woff += 32768 * 2;
  size_t big = (woff + 1048575) & ~(size_t)1048575;
  size_t SZ1 = (size_t)ROWS * G2D * 2;          // 50.3 MB each
  size_t SZH = (size_t)SEQ * MT * 128 * 2;      // 100.7 MB each
  unsigned short* bufA = (unsigned short*)(ws + big);
  unsigned short* bufB = (unsigned short*)(ws + big + SZ1);
  unsigned short* bufC = (unsigned short*)(ws + big + 2 * SZ1);
  unsigned short* bufD = (unsigned short*)(ws + big + 3 * SZ1);
  unsigned short* bufE = (unsigned short*)(ws + big + 3 * SZ1 + SZH);

  // adjacency prep
  k_deg<<<1, NSN, 0, stream>>>(adj, dinv);
  k_dad<<<NSN, NSN, 0, stream>>>(adj, dinv, dad);
  k_adp<<<NSN, NSN, 0, stream>>>(L1, L2, aadp);
  // weights -> bf16 (W2/W2_adp transposed for contiguous-K B operands)
  k_cvt_t64<<<16, 256, 0, stream>>>(W2,  W2t);
  k_cvt_t64<<<16, 256, 0, stream>>>(W2a, W2at);
  k_cvt<<<128, 256, 0, stream>>>(Wih0, Wih0b, 32768);
  k_cvt<<<128, 256, 0, stream>>>(Whh0, Whh0b, 32768);
  k_cvt<<<256, 256, 0, stream>>>(Wih1, Wih1b, 65536);
  k_cvt<<<128, 256, 0, stream>>>(Whh1, Whh1b, 32768);
  // GCN layer 1: bufA/bufB = transposed h@W panels, bufC = x1 [b][s][g]
  size_t tot = (size_t)ROWS * G2D;
  k_xw1<<<(unsigned)((tot + 255) / 256), 256, 0, stream>>>(x_f, x_o, W1, W1a, bufA, bufB);
  k_gcn_spmm<<<dim3(NSN / 16, BNM), 32, 0, stream>>>(dad, aadp, bufA, bufB, bufC);
  // GCN layer 2: reuse bufA/bufB, bufC overwritten with x2
  k_xw2<<<ROWS / 16, 32, 0, stream>>>(bufC, W2t, W2at, bufA, bufB);
  k_gcn_spmm<<<dim3(NSN / 16, BNM), 32, 0, stream>>>(dad, aadp, bufA, bufB, bufC);
  // BiLSTM layers (grid.y = direction)
  k_lstm<<<dim3(SEQ / 16, 2), 32, 0, stream>>>(bufC, Wih0b, Whh0b, bih0, bhh0, bufD, G2D, 0);
  k_lstm<<<dim3(SEQ / 16, 2), 32, 0, stream>>>(bufD, Wih1b, Whh1b, bih1, bhh1, bufE, 128, 1);
  // MLP + reductions
  k_mlp<<<(unsigned)(((size_t)NB * NSN * MT + 255) / 256), 256, 0, stream>>>(bufE, Wout, out);
  k_qhat<<<NB * MT, NSN, 0, stream>>>(out, slen, theta, out);
}
